// TitanArchitecture_38671885533815
// MI455X (gfx1250) — matrix operand order repaired and checked by reading, not compiled, not run on hardware
//
#include <hip/hip_runtime.h>
#include <math.h>

// ---------------- problem dims ----------------
#define B_ROWS   2048
#define H_ROWS   16384
#define E_ROWS   64
#define D_INDIM  384
#define KDIM     128
#define VDIM     128
#define HDIM     256
#define SPLIT_H  4
#define SEG_H    (H_ROWS / SPLIT_H)   // 4096
#define RB       (B_ROWS / 16)        // 128 query row-blocks

typedef __attribute__((ext_vector_type(16))) __bf16 v16bf;
typedef __attribute__((ext_vector_type(8)))  __bf16 v8bf;
typedef __attribute__((ext_vector_type(8)))  float  v8f;

// ---------------- bf16 split helpers (RNE) ----------------
__device__ __forceinline__ __bf16 f2bf(float f) {
  unsigned u = __float_as_uint(f);
  u += 0x7FFFu + ((u >> 16) & 1u);
  unsigned short s = (unsigned short)(u >> 16);
  return __builtin_bit_cast(__bf16, s);
}
__device__ __forceinline__ float bf2f(__bf16 b) {
  unsigned short s = __builtin_bit_cast(unsigned short, b);
  return __uint_as_float(((unsigned)s) << 16);
}

// ---------------- generic fp32 GEMM: C = A[MxK] @ W[NxK]^T (+bias, relu) ----
// Optional bf16 hi/lo split output for feeding the WMMA flash kernel.
__global__ void gemm_nt_kernel(const float* __restrict__ A, const float* __restrict__ W,
                               const float* __restrict__ bias, float* __restrict__ Cf,
                               __bf16* __restrict__ Chi, __bf16* __restrict__ Clo,
                               int M, int N, int K, int relu)
{
  const int idx = blockIdx.x * blockDim.x + threadIdx.x;
  if (idx >= M * N) return;
  const int m = idx / N, n = idx % N;
  float a = bias ? bias[n] : 0.0f;
  const float* ar = A + (size_t)m * K;
  const float* wr = W + (size_t)n * K;
  for (int k = 0; k < K; ++k) a = fmaf(ar[k], wr[k], a);
  if (relu) a = fmaxf(a, 0.0f);
  if (Cf) Cf[idx] = a;
  if (Chi) {
    __bf16 h = f2bf(a);
    Chi[idx] = h;
    if (Clo) Clo[idx] = f2bf(a - bf2f(h));
  }
}

// C[N1xN2] = A[MxN1]^T @ Bm[MxN2]   (grad outer products)
__global__ void gemm_tn_kernel(const float* __restrict__ A, const float* __restrict__ Bm,
                               float* __restrict__ C, int M, int N1, int N2)
{
  const int i = blockIdx.x * blockDim.x + threadIdx.x;
  if (i >= N1 * N2) return;
  const int n1 = i / N2, n2 = i % N2;
  float s = 0.0f;
  for (int m = 0; m < M; ++m) s = fmaf(A[(size_t)m * N1 + n1], Bm[(size_t)m * N2 + n2], s);
  C[i] = s;
}

__global__ void colsum_kernel(const float* __restrict__ A, float* __restrict__ out, int M, int N)
{
  const int n = blockIdx.x * blockDim.x + threadIdx.x;
  if (n >= N) return;
  float s = 0.0f;
  for (int m = 0; m < M; ++m) s += A[(size_t)m * N + n];
  out[n] = s;
}

// dout = 2*(pred - v) / (B*VD), in place over pred
__global__ void dout_kernel(float* __restrict__ pd, const float* __restrict__ v, int n)
{
  const int i = blockIdx.x * blockDim.x + threadIdx.x;
  if (i >= n) return;
  pd[i] = (pd[i] - v[i]) * 7.62939453125e-06f;   // 2/262144
}

// dh = (dout @ mW2) * relu'(h)
__global__ void dh_kernel(const float* __restrict__ dout, const float* __restrict__ mW2,
                          const float* __restrict__ h, float* __restrict__ dh)
{
  const int i = blockIdx.x * blockDim.x + threadIdx.x;
  if (i >= B_ROWS * HDIM) return;
  const int b = i >> 8, hd = i & 255;
  float s = 0.0f;
  for (int vd = 0; vd < VDIM; ++vd) s = fmaf(dout[b * VDIM + vd], mW2[vd * HDIM + hd], s);
  dh[i] = h[i] > 0.0f ? s : 0.0f;
}

// deterministic fixed-order squared-sum reduction (no float atomics)
__global__ void sq_partial_kernel(const float* g1, int n1, const float* g2, int n2,
                                  const float* g3, int n3, const float* g4, int n4,
                                  float* __restrict__ part)
{
  __shared__ float red[256];
  const int tid = threadIdx.x;
  const int gid = blockIdx.x * 256 + tid;
  const int stride = gridDim.x * 256;
  float s = 0.0f;
  for (int i = gid; i < n1; i += stride) { float x = g1[i]; s = fmaf(x, x, s); }
  for (int i = gid; i < n2; i += stride) { float x = g2[i]; s = fmaf(x, x, s); }
  for (int i = gid; i < n3; i += stride) { float x = g3[i]; s = fmaf(x, x, s); }
  for (int i = gid; i < n4; i += stride) { float x = g4[i]; s = fmaf(x, x, s); }
  red[tid] = s; __syncthreads();
  for (int o = 128; o > 0; o >>= 1) { if (tid < o) red[tid] += red[tid + o]; __syncthreads(); }
  if (tid == 0) part[blockIdx.x] = red[0];
}

__global__ void sq_final_kernel(const float* __restrict__ part, int nb, float* __restrict__ out)
{
  __shared__ float red[256];
  const int tid = threadIdx.x;
  float s = 0.0f;
  for (int i = tid; i < nb; i += 256) s += part[i];
  red[tid] = s; __syncthreads();
  for (int o = 128; o > 0; o >>= 1) { if (tid < o) red[tid] += red[tid + o]; __syncthreads(); }
  if (tid == 0) out[0] = red[0];
}

// new_p = 0.9*p + 0.9*buf - eff_theta*g  across the four param tensors
__global__ void newparams_kernel(const float* __restrict__ tot,
    const float* mW1, const float* bufW1, const float* gW1,
    const float* mb1, const float* bufb1, const float* gb1,
    const float* mW2, const float* bufW2, const float* gW2,
    const float* mb2, const float* bufb2, const float* gb2,
    float* nW1, float* nb1, float* nW2, float* nb2)
{
  const float avg = sqrtf(tot[0] * 0.25f);
  const float eff = 0.1f / (1.0f + __expf(-avg));
  const int i = blockIdx.x * blockDim.x + threadIdx.x;
  if (i < 32768)       { nW1[i] = 0.9f * mW1[i] + 0.9f * bufW1[i] - eff * gW1[i]; }
  else if (i < 33024)  { int j = i - 32768; nb1[j] = 0.9f * mb1[j] + 0.9f * bufb1[j] - eff * gb1[j]; }
  else if (i < 65792)  { int j = i - 33024; nW2[j] = 0.9f * mW2[j] + 0.9f * bufW2[j] - eff * gW2[j]; }
  else if (i < 65920)  { int j = i - 65792; nb2[j] = 0.9f * mb2[j] + 0.9f * bufb2[j] - eff * gb2[j]; }
}

// po += kvec (broadcast); gate = sigmoid(po . Wg + bg)
__global__ void pogate_kernel(float* __restrict__ po, const float* __restrict__ kvec,
                              const float* __restrict__ Wg, const float* __restrict__ bg,
                              float* __restrict__ gate)
{
  const int b = blockIdx.x * blockDim.x + threadIdx.x;
  if (b >= B_ROWS) return;
  float s = bg[0];
  for (int v = 0; v < VDIM; ++v) {
    float t = po[(size_t)b * VDIM + v] + kvec[v];
    po[(size_t)b * VDIM + v] = t;
    s = fmaf(t, Wg[v], s);
  }
  gate[b] = 1.0f / (1.0f + __expf(-s));
}

// transposed bf16 convert: dstT[c*rows + r] = bf16(src[r*cols + c])
// (gives P@V B-fragments a contiguous 32B v16bf load instead of a 16-way gather)
__global__ void cvt_bf16_t_kernel(const float* __restrict__ src, __bf16* __restrict__ dstT,
                                  int rows, int cols)
{
  const int o = blockIdx.x * blockDim.x + threadIdx.x;
  if (o >= rows * cols) return;
  const int c = o / rows, r = o % rows;
  dstT[o] = f2bf(src[(size_t)r * cols + c]);
}

// ---------------- flash attention (WMMA bf16x3 logits, bf16x2 P@V) --------
// One wave (32 lanes) per (16 query rows) x (H/SPLIT_H segment).
// Fragment layouts used here (contiguous K per lane half; C layout as in CDNA5 ISA 7.12.2):
//   A 16x32 bf16 : lane m=L&15; elems 0..15 -> K=koffA+i,
//                  koffA = (L>=16)?16:0
//   B 32x16 bf16 : lane n=L&15; elems 0..15 -> K=koffB+i, koffB = (L>=16)?16:0
//   C 16x16 f32  : lane n=L&15; elem r -> row r + ((L>=16)?8:0)
// kh is row-major along K (contiguous B-frag loads); history is pre-transposed
// [d][j] so the P@V B-frag is also one contiguous 32B load.
// Streamed B-fragments are double-buffered so iteration i+1's loads issue
// before iteration i's WMMAs (avoids s_wait_loadcnt 0 after every own load).
__global__ __launch_bounds__(32, 1) void flash_attn_kernel(
    const __bf16* __restrict__ qhi, const __bf16* __restrict__ qlo,
    const __bf16* __restrict__ khi, const __bf16* __restrict__ klo,
    const __bf16* __restrict__ vT,
    float* __restrict__ pacc, float* __restrict__ pm, float* __restrict__ pl)
{
  __shared__ alignas(32) float  sS[16 * 32];
  __shared__ alignas(32) __bf16 sPhi[16 * 32];
  __shared__ alignas(32) __bf16 sPlo[16 * 32];
  __shared__ alignas(32) float  sAlpha[16];

  const int lane  = threadIdx.x;
  const int rb    = blockIdx.x;
  const int seg   = blockIdx.y;
  const int ln15  = lane & 15;
  const int half  = lane >> 4;
  const int koffA = half * 16;
  const int koffB = half * 16;
  const int rbase = half * 8;

  // ---- preload Q A-fragments (hi/lo) for all 12 K-steps: stays in VGPRs ----
  v16bf qh[12], ql[12];
  {
    const __bf16* qr_hi = qhi + (size_t)(rb * 16 + ln15) * D_INDIM;
    const __bf16* qr_lo = qlo + (size_t)(rb * 16 + ln15) * D_INDIM;
#pragma unroll
    for (int kk = 0; kk < 12; ++kk) {
      const int k0 = kk * 32;
      v8bf a0 = *(const v8bf*)(qr_hi + k0 + koffA);
      v8bf a1 = *(const v8bf*)(qr_hi + k0 + 8 + koffA);
      qh[kk] = __builtin_shufflevector(a0, a1, 0,1,2,3,4,5,6,7,8,9,10,11,12,13,14,15);
      v8bf b0 = *(const v8bf*)(qr_lo + k0 + koffA);
      v8bf b1 = *(const v8bf*)(qr_lo + k0 + 8 + koffA);
      ql[kk] = __builtin_shufflevector(b0, b1, 0,1,2,3,4,5,6,7,8,9,10,11,12,13,14,15);
    }
  }

  // fp32 accumulator: 16 rows x 384 cols = 24 C-fragments (192 VGPRs)
  v8f acc[24];
#pragma unroll
  for (int i = 0; i < 24; ++i)
#pragma unroll
    for (int e = 0; e < 8; ++e) acc[i][e] = 0.0f;

  float mrow = -3.0e38f;   // per-row running max (lanes 0..15)
  float lrow = 0.0f;       // per-row running sum-exp (lanes 0..15)

  const int jend = seg * SEG_H + SEG_H;
  for (int j0 = seg * SEG_H; j0 < jend; j0 += 32) {
    if (j0 + 32 < jend) {   // CDNA5 global_prefetch_b8 for next chunk (L2-resident data)
      __builtin_prefetch((const void*)(khi + (size_t)(j0 + 32 + lane) * D_INDIM), 0, 1);
      __builtin_prefetch((const void*)(vT + (size_t)(lane * 16 + half) * H_ROWS + j0 + 32), 0, 1);
    }

    // ---- S = Q @ kh^T for 32 history cols: two 16x16 tiles, bf16x3 split ----
#pragma unroll
    for (int t = 0; t < 2; ++t) {
      v8f c;
#pragma unroll
      for (int e = 0; e < 8; ++e) c[e] = 0.0f;
      const __bf16* kr_hi = khi + (size_t)(j0 + t * 16 + ln15) * D_INDIM;
      const __bf16* kr_lo = klo + (size_t)(j0 + t * 16 + ln15) * D_INDIM;
      // software pipeline: fetch kk+1 fragments before kk's WMMAs
      v16bf bh_c = *(const v16bf*)(kr_hi + koffB);
      v16bf bl_c = *(const v16bf*)(kr_lo + koffB);
#pragma unroll
      for (int kk = 0; kk < 12; ++kk) {
        v16bf bh_n, bl_n;
        if (kk < 11) {
          const int k1 = (kk + 1) * 32;
          bh_n = *(const v16bf*)(kr_hi + k1 + koffB);
          bl_n = *(const v16bf*)(kr_lo + k1 + koffB);
        }
        c = __builtin_amdgcn_wmma_f32_16x16x32_bf16(false, qh[kk], false, bh_c, (short)0, c, false, false);
        c = __builtin_amdgcn_wmma_f32_16x16x32_bf16(false, qh[kk], false, bl_c, (short)0, c, false, false);
        c = __builtin_amdgcn_wmma_f32_16x16x32_bf16(false, ql[kk], false, bh_c, (short)0, c, false, false);
        if (kk < 11) { bh_c = bh_n; bl_c = bl_n; }
      }
#pragma unroll
      for (int e = 0; e < 8; ++e) sS[(rbase + e) * 32 + t * 16 + ln15] = c[e];
    }
    __syncthreads();

    // ---- online softmax: lane r < 16 owns row r ----
    if (lane < 16) {
      float s[32];
      float mx = mrow;
#pragma unroll
      for (int j = 0; j < 32; ++j) { s[j] = sS[ln15 * 32 + j]; mx = fmaxf(mx, s[j]); }
      const float al = __expf(mrow - mx);
      float sum = 0.0f;
#pragma unroll
      for (int j = 0; j < 32; ++j) {
        const float p = __expf(s[j] - mx);
        sum += p;
        const __bf16 ph = f2bf(p);
        sPhi[ln15 * 32 + j] = ph;
        sPlo[ln15 * 32 + j] = f2bf(p - bf2f(ph));
      }
      lrow = al * lrow + sum;
      mrow = mx;
      sAlpha[ln15] = al;
    }
    __syncthreads();

    // ---- rescale accumulator by per-row alpha ----
    float al8[8];
#pragma unroll
    for (int e = 0; e < 8; ++e) al8[e] = sAlpha[rbase + e];
#pragma unroll
    for (int dt = 0; dt < 24; ++dt)
#pragma unroll
      for (int e = 0; e < 8; ++e) acc[dt][e] *= al8[e];

    // ---- P A-fragments from LDS (C-layout -> A-layout transpose) ----
    v16bf pAh, pAl;
    {
      v8bf x0 = *(const v8bf*)(&sPhi[ln15 * 32 + koffA]);
      v8bf x1 = *(const v8bf*)(&sPhi[ln15 * 32 + 8 + koffA]);
      pAh = __builtin_shufflevector(x0, x1, 0,1,2,3,4,5,6,7,8,9,10,11,12,13,14,15);
      v8bf y0 = *(const v8bf*)(&sPlo[ln15 * 32 + koffA]);
      v8bf y1 = *(const v8bf*)(&sPlo[ln15 * 32 + 8 + koffA]);
      pAl = __builtin_shufflevector(y0, y1, 0,1,2,3,4,5,6,7,8,9,10,11,12,13,14,15);
    }

    // ---- acc += P @ V over all 24 d-tiles (bf16x2), double-buffered vT ----
    {
      const __bf16* vbase = vT + (size_t)ln15 * H_ROWS + j0 + koffB;
      v16bf vb_c = *(const v16bf*)vbase;
#pragma unroll
      for (int dt = 0; dt < 24; ++dt) {
        v16bf vb_n;
        if (dt < 23)
          vb_n = *(const v16bf*)(vbase + (size_t)(dt + 1) * 16 * H_ROWS);
        acc[dt] = __builtin_amdgcn_wmma_f32_16x16x32_bf16(false, pAh, false, vb_c, (short)0, acc[dt], false, false);
        acc[dt] = __builtin_amdgcn_wmma_f32_16x16x32_bf16(false, pAl, false, vb_c, (short)0, acc[dt], false, false);
        if (dt < 23) vb_c = vb_n;
      }
    }
    __syncthreads();   // protect sS / sP reuse next chunk
  }

  // ---- write segment partials ----
  const int pidx = rb * SPLIT_H + seg;
  if (lane < 16) { pm[pidx * 16 + ln15] = mrow; pl[pidx * 16 + ln15] = lrow; }
#pragma unroll
  for (int dt = 0; dt < 24; ++dt)
#pragma unroll
    for (int e = 0; e < 8; ++e)
      pacc[((size_t)pidx * 16 + rbase + e) * D_INDIM + dt * 16 + ln15] = acc[dt][e];
}

// merge SPLIT_H softmax segments: ctx = sum_s w_s*acc_s / sum_s w_s*l_s
__global__ void attn_combine_kernel(const float* __restrict__ pacc, const float* __restrict__ pm,
                                    const float* __restrict__ pl, float* __restrict__ ctx)
{
  const int b = blockIdx.x;
  const int rb = b >> 4, r = b & 15;
  float M = -3.0e38f;
#pragma unroll
  for (int s = 0; s < SPLIT_H; ++s) M = fmaxf(M, pm[(rb * SPLIT_H + s) * 16 + r]);
  float w[SPLIT_H];
  float L = 0.0f;
#pragma unroll
  for (int s = 0; s < SPLIT_H; ++s) {
    w[s] = __expf(pm[(rb * SPLIT_H + s) * 16 + r] - M);
    L += w[s] * pl[(rb * SPLIT_H + s) * 16 + r];
  }
  const float invL = 1.0f / L;
  for (int d = threadIdx.x; d < D_INDIM; d += blockDim.x) {
    float a = 0.0f;
#pragma unroll
    for (int s = 0; s < SPLIT_H; ++s)
      a += w[s] * pacc[((size_t)(rb * SPLIT_H + s) * 16 + r) * D_INDIM + d];
    ctx[(size_t)b * D_INDIM + d] = a * invL;
  }
}

// entity path: avg over E rows -> ec = avg@Wep^T+bep -> eg = sigmoid(ec.Weg+beg)
__global__ void entity_kernel(const float* __restrict__ ent, const float* __restrict__ Wep,
                              const float* __restrict__ bep, const float* __restrict__ Weg,
                              const float* __restrict__ beg, float* __restrict__ ec,
                              float* __restrict__ eg)
{
  __shared__ float avg[D_INDIM];
  __shared__ float ecs[VDIM];
  const int t = threadIdx.x;   // blockDim = 128
  for (int d = t; d < D_INDIM; d += 128) {
    float s = 0.0f;
    for (int e = 0; e < E_ROWS; ++e) s += ent[(size_t)e * D_INDIM + d];
    avg[d] = s * (1.0f / 64.0f);
  }
  __syncthreads();
  {
    float s = bep[t];
    for (int d = 0; d < D_INDIM; ++d) s = fmaf(avg[d], Wep[(size_t)t * D_INDIM + d], s);
    ec[t] = s; ecs[t] = s;
  }
  __syncthreads();
  if (t == 0) {
    float s = beg[0];
    for (int v = 0; v < VDIM; ++v) s = fmaf(ecs[v], Weg[v], s);
    eg[0] = 1.0f / (1.0f + __expf(-s));
  }
}

__global__ void fuse_kernel(const float* __restrict__ po, const float* __restrict__ st,
                            const float* __restrict__ gate, const float* __restrict__ ec,
                            const float* __restrict__ eg, float* __restrict__ out)
{
  const int i = blockIdx.x * blockDim.x + threadIdx.x;
  if (i >= B_ROWS * VDIM) return;
  const int b = i >> 7, vd = i & 127;
  const float g = gate[b];
  const float f = g * po[i] + (1.0f - g) * st[i];
  const float e = eg[0];
  out[i] = e * ec[vd] + (1.0f - e) * f;
}

// ----------------------------- host driver --------------------------------
extern "C" void kernel_launch(void* const* d_in, const int* in_sizes, int n_in,
                              void* d_out, int out_size, void* d_ws, size_t ws_size,
                              hipStream_t stream)
{
  const float* x     = (const float*)d_in[0];
  const float* hist  = (const float*)d_in[1];
  const float* ent   = (const float*)d_in[2];
  const float* W_K   = (const float*)d_in[3];
  const float* W_V   = (const float*)d_in[4];
  const float* mW1   = (const float*)d_in[5];
  const float* mb1   = (const float*)d_in[6];
  const float* mW2   = (const float*)d_in[7];
  const float* mb2   = (const float*)d_in[8];
  const float* bufW1 = (const float*)d_in[9];
  const float* bufb1 = (const float*)d_in[10];
  const float* bufW2 = (const float*)d_in[11];
  const float* bufb2 = (const float*)d_in[12];
  const float* kvec  = (const float*)d_in[13];
  const float* Wq    = (const float*)d_in[14];
  const float* Wk    = (const float*)d_in[15];
  const float* Wst   = (const float*)d_in[16];
  const float* bst   = (const float*)d_in[17];
  const float* Wg    = (const float*)d_in[18];
  const float* bg    = (const float*)d_in[19];
  // d_in[20..23] (Wgt,bgt,Wip,bip) feed only gating_loss -> dead for the output
  const float* Wep   = (const float*)d_in[24];
  const float* bep   = (const float*)d_in[25];
  const float* Weg   = (const float*)d_in[26];
  const float* beg   = (const float*)d_in[27];
  (void)in_sizes; (void)n_in; (void)out_size; (void)ws_size;

  // bump allocator over workspace (total ~65 MB)
  char* base = (char*)d_ws;
  size_t off = 0;
  auto alloc = [&](size_t bytes) -> char* {
    off = (off + 255) & ~(size_t)255;
    char* p = base + off;
    off += bytes;
    return p;
  };

  float* kbuf = (float*)alloc((size_t)B_ROWS * KDIM * 4);
  float* vbuf = (float*)alloc((size_t)B_ROWS * VDIM * 4);   // reused as persistent_out
  float* hbuf = (float*)alloc((size_t)B_ROWS * HDIM * 4);
  float* dout = (float*)alloc((size_t)B_ROWS * VDIM * 4);   // pred, then dL/dpred in place
  float* dhb  = (float*)alloc((size_t)B_ROWS * HDIM * 4);   // reused as h2
  float* gW1  = (float*)alloc((size_t)HDIM * KDIM * 4);
  float* gb1  = (float*)alloc((size_t)HDIM * 4);
  float* gW2  = (float*)alloc((size_t)VDIM * HDIM * 4);
  float* gb2  = (float*)alloc((size_t)VDIM * 4);
  float* redp = (float*)alloc(64 * 4);
  float* tots = (float*)alloc(4);
  float* nW1  = (float*)alloc((size_t)HDIM * KDIM * 4);
  float* nb1  = (float*)alloc((size_t)HDIM * 4);
  float* nW2  = (float*)alloc((size_t)VDIM * HDIM * 4);
  float* nb2  = (float*)alloc((size_t)VDIM * 4);
  float* gate = (float*)alloc((size_t)B_ROWS * 4);
  float* ctx  = (float*)alloc((size_t)B_ROWS * D_INDIM * 4);
  float* stb  = (float*)alloc((size_t)B_ROWS * VDIM * 4);
  float* ecb  = (float*)alloc((size_t)VDIM * 4);
  float* egb  = (float*)alloc(4);
  __bf16* qhi  = (__bf16*)alloc((size_t)B_ROWS * D_INDIM * 2);
  __bf16* qlo  = (__bf16*)alloc((size_t)B_ROWS * D_INDIM * 2);
  __bf16* khhi = (__bf16*)alloc((size_t)H_ROWS * D_INDIM * 2);
  __bf16* khlo = (__bf16*)alloc((size_t)H_ROWS * D_INDIM * 2);
  __bf16* vTb  = (__bf16*)alloc((size_t)H_ROWS * D_INDIM * 2);   // history^T, bf16 hi
  float* pacc = (float*)alloc((size_t)RB * SPLIT_H * 16 * D_INDIM * 4);
  float* pmb  = (float*)alloc((size_t)RB * SPLIT_H * 16 * 4);
  float* plb  = (float*)alloc((size_t)RB * SPLIT_H * 16 * 4);

  float* po = vbuf;   // safe sequential reuse: v consumed at step 5, po written at 15
  float* h2 = dhb;    // dh consumed at steps 9/10, h2 written at 14

  const int T = 256;
  auto nb = [](long n, int t) { return (int)((n + t - 1) / t); };

  // ---- Titan memory update (exact fp32) ----
  gemm_nt_kernel<<<nb((long)B_ROWS * KDIM, T), T, 0, stream>>>(x, W_K, nullptr, kbuf, nullptr, nullptr, B_ROWS, KDIM, D_INDIM, 0);
  gemm_nt_kernel<<<nb((long)B_ROWS * VDIM, T), T, 0, stream>>>(x, W_V, nullptr, vbuf, nullptr, nullptr, B_ROWS, VDIM, D_INDIM, 0);
  gemm_nt_kernel<<<nb((long)B_ROWS * HDIM, T), T, 0, stream>>>(kbuf, mW1, mb1, hbuf, nullptr, nullptr, B_ROWS, HDIM, KDIM, 1);
  gemm_nt_kernel<<<nb((long)B_ROWS * VDIM, T), T, 0, stream>>>(hbuf, mW2, mb2, dout, nullptr, nullptr, B_ROWS, VDIM, HDIM, 0);
  dout_kernel<<<nb((long)B_ROWS * VDIM, T), T, 0, stream>>>(dout, vbuf, B_ROWS * VDIM);
  gemm_tn_kernel<<<nb((long)VDIM * HDIM, T), T, 0, stream>>>(dout, hbuf, gW2, B_ROWS, VDIM, HDIM);
  colsum_kernel<<<nb(VDIM, T), T, 0, stream>>>(dout, gb2, B_ROWS, VDIM);
  dh_kernel<<<nb((long)B_ROWS * HDIM, T), T, 0, stream>>>(dout, mW2, hbuf, dhb);
  gemm_tn_kernel<<<nb((long)HDIM * KDIM, T), T, 0, stream>>>(dhb, kbuf, gW1, B_ROWS, HDIM, KDIM);
  colsum_kernel<<<nb(HDIM, T), T, 0, stream>>>(dhb, gb1, B_ROWS, HDIM);
  sq_partial_kernel<<<64, 256, 0, stream>>>(gW1, HDIM * KDIM, gb1, HDIM, gW2, VDIM * HDIM, gb2, VDIM, redp);
  sq_final_kernel<<<1, 256, 0, stream>>>(redp, 64, tots);
  newparams_kernel<<<nb(65920, T), T, 0, stream>>>(tots, mW1, bufW1, gW1, mb1, bufb1, gb1,
                                                   mW2, bufW2, gW2, mb2, bufb2, gb2,
                                                   nW1, nb1, nW2, nb2);
  gemm_nt_kernel<<<nb((long)B_ROWS * HDIM, T), T, 0, stream>>>(kbuf, nW1, nb1, h2, nullptr, nullptr, B_ROWS, HDIM, KDIM, 1);
  gemm_nt_kernel<<<nb((long)B_ROWS * VDIM, T), T, 0, stream>>>(h2, nW2, nb2, po, nullptr, nullptr, B_ROWS, VDIM, HDIM, 0);
  pogate_kernel<<<nb(B_ROWS, T), T, 0, stream>>>(po, kvec, Wg, bg, gate);

  // ---- attention projections: write bf16 hi/lo splits directly ----
  gemm_nt_kernel<<<nb((long)B_ROWS * D_INDIM, T), T, 0, stream>>>(x, Wq, nullptr, nullptr, qhi, qlo, B_ROWS, D_INDIM, D_INDIM, 0);
  gemm_nt_kernel<<<nb((long)H_ROWS * D_INDIM, T), T, 0, stream>>>(hist, Wk, nullptr, nullptr, khhi, khlo, H_ROWS, D_INDIM, D_INDIM, 0);
  cvt_bf16_t_kernel<<<nb((long)H_ROWS * D_INDIM, T), T, 0, stream>>>(hist, vTb, H_ROWS, D_INDIM);

  // ---- WMMA flash attention over H, segmented for occupancy ----
  flash_attn_kernel<<<dim3(RB, SPLIT_H), 32, 0, stream>>>(qhi, qlo, khhi, khlo, vTb, pacc, pmb, plb);
  attn_combine_kernel<<<B_ROWS, 128, 0, stream>>>(pacc, pmb, plb, ctx);
  gemm_nt_kernel<<<nb((long)B_ROWS * VDIM, T), T, 0, stream>>>(ctx, Wst, bst, stb, nullptr, nullptr, B_ROWS, VDIM, D_INDIM, 0);

  // ---- entity path + final fusion ----
  entity_kernel<<<1, 128, 0, stream>>>(ent, Wep, bep, Weg, beg, ecb, egb);
  fuse_kernel<<<nb((long)B_ROWS * VDIM, T), T, 0, stream>>>(po, stb, gate, ecb, egb, (float*)d_out);
}
